// Quantizer_9431748182641
// MI455X (gfx1250) — matrix operand order repaired and checked by reading, not compiled, not run on hardware
//
#include <hip/hip_runtime.h>
#include <hip/hip_bf16.h>

// CDNA5 (gfx1250) WMMA types
typedef __attribute__((ext_vector_type(16))) __bf16 v16bf;
typedef __attribute__((ext_vector_type(8)))  float  v8f;

#if defined(__has_builtin)
# if __has_builtin(__builtin_amdgcn_global_load_async_to_lds_b128) && \
     __has_builtin(__builtin_amdgcn_s_wait_asynccnt)
#  define VQ_USE_ASYNC 1
# endif
# if __has_builtin(__builtin_amdgcn_sched_barrier)
#  define VQ_SCHED_FENCE() __builtin_amdgcn_sched_barrier(0)
# endif
#endif
#ifndef VQ_USE_ASYNC
# define VQ_USE_ASYNC 0
#endif
#ifndef VQ_SCHED_FENCE
# define VQ_SCHED_FENCE()
#endif

#define TOK   32768          // B tokens
#define NCB   4096           // codebook entries
#define DIM   256            // code dim (K)
#define NTILES (NCB / 16)    // 256 n-tiles of 16 columns
#define NKF    (DIM / 32)    // 8 K-fragments of 32 per WMMA
#define ROWS_PER_BLOCK 128   // 8 waves x 16 rows
#define THREADS 256

#if VQ_USE_ASYNC
// builtin signature (from hipcc diagnostic): arg0 is int4* in AS1
typedef int vq_int4 __attribute__((__vector_size__(4 * sizeof(int))));
typedef __attribute__((address_space(1))) vq_int4* vq_g4p;
typedef __attribute__((address_space(3))) vq_int4* vq_l4p;
#endif

// ---------------------------------------------------------------------------
// Kernel 1: normalize codebook rows, convert to bf16, pre-pack into the
// WMMA B-matrix (32x16, K x N) per-lane layout so the GEMM kernel can load
// fragments as contiguous 32B per lane.
//   thread t = kf*32 + L ; lane L: column N = L&15, khalf = L>>4
//   element j (0..15): k = kf*32 + khalf*16 + j
// ---------------------------------------------------------------------------
__global__ __launch_bounds__(THREADS) void vq_pack_kernel(
    const float* __restrict__ W, __bf16* __restrict__ Wpack)
{
  __shared__ float psum[16][17];
  __shared__ float rscale[16];
  const int tid = threadIdx.x;
  const int nt  = blockIdx.x;

  // row-norm partial sums: 16 threads per row, 16 elements each
  {
    const int r = tid >> 4, cs = tid & 15;
    const float* p = W + ((size_t)nt * 16 + r) * DIM + cs * 16;
    float s = 0.f;
#pragma unroll
    for (int j = 0; j < 16; ++j) s += p[j] * p[j];
    psum[r][cs] = s;
  }
  __syncthreads();
  if (tid < 16) {
    float t = 0.f;
#pragma unroll
    for (int j = 0; j < 16; ++j) t += psum[tid][j];
    const float nrm = sqrtf(t);
    rscale[tid] = 1.0f / fmaxf(nrm, 1e-12f);   // matches F.normalize eps
  }
  __syncthreads();

  const int kf    = tid >> 5;
  const int L     = tid & 31;
  const int nloc  = L & 15;
  const int khalf = L >> 4;
  const float* wr = W + ((size_t)nt * 16 + nloc) * DIM + kf * 32 + khalf * 16;
  const float sc  = rscale[nloc];
  v16bf pv;
#pragma unroll
  for (int j = 0; j < 16; ++j) pv[j] = (__bf16)(wr[j] * sc);
  *(v16bf*)(Wpack + (size_t)nt * (NKF * 512) + tid * 16) = pv;
}

// ---------------------------------------------------------------------------
// Kernel 2: sim = E_bf16 @ Wn_bf16^T via v_wmma_f32_16x16x32_bf16,
// streaming running argmax per row; then gather e_q = W[idx] and per-block
// squared-diff partial sum. W tiles double-buffered through LDS using
// CDNA5 async global->LDS copies when available.
// ---------------------------------------------------------------------------
__global__ __launch_bounds__(THREADS) void vq_sim_argmax_kernel(
    const float* __restrict__ e,
    const float* __restrict__ W,
    const __bf16* __restrict__ Wpack,
    float* __restrict__ out_eq,
    float* __restrict__ partials)
{
  __shared__ __align__(32) __bf16 lds_b[2][NKF * 512];  // 2 x 8KB tiles
  __shared__ float lds_max[8][16][16];
  __shared__ int   lds_idx[8][16][16];
  __shared__ int   row_arg[ROWS_PER_BLOCK];
  __shared__ float redbuf[THREADS];

  const int tid   = threadIdx.x;
  const int wave  = tid >> 5;
  const int lane  = tid & 31;
  const int khalf = lane >> 4;
  const int nlan  = lane & 15;

  const int m0  = blockIdx.x * ROWS_PER_BLOCK + wave * 16;
  const int row = m0 + nlan;

  // A fragments: 16x32 bf16 per K-fragment, built on the fly from f32 E.
  // layout (matches packed B): lanes 0-15 -> K {0..15}; lanes 16-31 -> K {16..31}
  v16bf afrag[NKF];
#pragma unroll
  for (int kf = 0; kf < NKF; ++kf) {
    const float* p = e + (size_t)row * DIM + kf * 32 + khalf * 16;
    v16bf a;
#pragma unroll
    for (int j = 0; j < 8; ++j) a[j]     = (__bf16)p[j];
#pragma unroll
    for (int j = 0; j < 8; ++j) a[8 + j] = (__bf16)p[8 + j];
    afrag[kf] = a;
  }

  float rmax[8];
  int   rarg[8];
#pragma unroll
  for (int v = 0; v < 8; ++v) { rmax[v] = -3.402823466e38f; rarg[v] = 0; }

  // prologue: stage n-tile 0 (each thread copies 32B)
#if VQ_USE_ASYNC
  {
    const char* g = (const char*)Wpack + (size_t)tid * 32;
    char*       l = (char*)&lds_b[0][0] + (size_t)tid * 32;
    __builtin_amdgcn_global_load_async_to_lds_b128((vq_g4p)g,        (vq_l4p)l,        0, 0);
    __builtin_amdgcn_global_load_async_to_lds_b128((vq_g4p)(g + 16), (vq_l4p)(l + 16), 0, 0);
    __builtin_amdgcn_s_wait_asynccnt(0);
  }
  __syncthreads();
#else
  {
    const uint4* s = (const uint4*)(Wpack + (size_t)tid * 16);
    uint4 a0 = s[0], a1 = s[1];
    uint4* d = (uint4*)(&lds_b[0][tid * 16]);
    d[0] = a0; d[1] = a1;
  }
  __syncthreads();
#endif

  for (int nt = 0; nt < NTILES; ++nt) {
    const int  buf  = nt & 1;
    const bool more = (nt + 1 < NTILES);

#if VQ_USE_ASYNC
    // issue async copy of next tile FIRST; 8 WMMAs below hide the latency
    if (more) {
      const char* g = (const char*)(Wpack + (size_t)(nt + 1) * (NKF * 512)) + (size_t)tid * 32;
      char*       l = (char*)&lds_b[buf ^ 1][0] + (size_t)tid * 32;
      __builtin_amdgcn_global_load_async_to_lds_b128((vq_g4p)g,        (vq_l4p)l,        0, 0);
      __builtin_amdgcn_global_load_async_to_lds_b128((vq_g4p)(g + 16), (vq_l4p)(l + 16), 0, 0);
    }
    VQ_SCHED_FENCE();
#else
    uint4 nb0, nb1;
    if (more) {
      const uint4* s = (const uint4*)(Wpack + (size_t)(nt + 1) * (NKF * 512) + tid * 16);
      nb0 = s[0]; nb1 = s[1];
    }
    VQ_SCHED_FENCE();  // pin the next-tile loads above the math
#endif

    // hoist ALL B-fragment LDS loads, then chain the WMMAs (partial dscnt
    // waits instead of one wait-0 stall per fragment)
    v16bf bfr[NKF];
#pragma unroll
    for (int kf = 0; kf < NKF; ++kf)
      bfr[kf] = *(const v16bf*)(&lds_b[buf][kf * 512 + lane * 16]);

    v8f c = {};
#pragma unroll
    for (int kf = 0; kf < NKF; ++kf)
      c = __builtin_amdgcn_wmma_f32_16x16x32_bf16(
              /*neg_a=*/false, afrag[kf], /*neg_b=*/false, bfr[kf],
              /*c_mod=*/(short)0, c, /*reuse_a=*/false, /*reuse_b=*/false);

    // C layout: VGPR v, lane L -> sim[row = v + 8*khalf][col = nt*16 + nlan]
    const int ncol = nt * 16 + nlan;
#pragma unroll
    for (int v = 0; v < 8; ++v) {
      const float x = c[v];
      if (x > rmax[v]) { rmax[v] = x; rarg[v] = ncol; }
    }

    if (more) {
#if VQ_USE_ASYNC
      __builtin_amdgcn_s_wait_asynccnt(0);
      __syncthreads();
#else
      uint4* d = (uint4*)(&lds_b[buf ^ 1][tid * 16]);
      d[0] = nb0; d[1] = nb1;
      __syncthreads();     // one barrier per iteration (double buffer)
#endif
    }
  }

  // cross-lane argmax reduction (same-wave LDS, in-order)
#pragma unroll
  for (int v = 0; v < 8; ++v) {
    lds_max[wave][v + 8 * khalf][nlan] = rmax[v];
    lds_idx[wave][v + 8 * khalf][nlan] = rarg[v];
  }
  if (lane < 16) {
    float m = -3.402823466e38f;
    int   a = 0;
#pragma unroll
    for (int j = 0; j < 16; ++j) {
      const float x = lds_max[wave][lane][j];
      if (x > m) { m = x; a = lds_idx[wave][lane][j]; }
    }
    row_arg[wave * 16 + lane] = a;
  }
  __syncthreads();

  // epilogue: e_q = W[idx] gather + squared-diff partial
  float lsum = 0.0f;
  const int base = blockIdx.x * ROWS_PER_BLOCK;
  for (int i = tid; i < ROWS_PER_BLOCK * DIM; i += THREADS) {
    const int r = i >> 8;            // DIM == 256
    const int d = i & (DIM - 1);
    const size_t gi = (size_t)(base + r) * DIM + d;
    const float w  = W[(size_t)row_arg[r] * DIM + d];
    const float ev = e[gi];
    out_eq[gi] = w;
    const float df = w - ev;
    lsum += df * df;
  }
  redbuf[tid] = lsum;
  __syncthreads();
  for (int s = THREADS / 2; s > 0; s >>= 1) {
    if (tid < s) redbuf[tid] += redbuf[tid + s];
    __syncthreads();
  }
  if (tid == 0) partials[blockIdx.x] = redbuf[0];
}

// ---------------------------------------------------------------------------
// Kernel 3: reduce 256 per-block partials -> loss = sum / (B*D)
// ---------------------------------------------------------------------------
__global__ __launch_bounds__(THREADS) void vq_finalize_kernel(
    const float* __restrict__ partials, float* __restrict__ loss)
{
  __shared__ float buf[THREADS];
  const int tid = threadIdx.x;
  buf[tid] = partials[tid];
  __syncthreads();
  for (int s = THREADS / 2; s > 0; s >>= 1) {
    if (tid < s) buf[tid] += buf[tid + s];
    __syncthreads();
  }
  if (tid == 0) loss[0] = buf[0] * (1.0f / ((float)TOK * (float)DIM));
}

// ---------------------------------------------------------------------------
extern "C" void kernel_launch(void* const* d_in, const int* in_sizes, int n_in,
                              void* d_out, int out_size, void* d_ws, size_t ws_size,
                              hipStream_t stream) {
  const float* e = (const float*)d_in[0];   // [TOK, DIM] f32
  const float* W = (const float*)d_in[1];   // [NCB, DIM] f32
  // d_in[2] = return_assignment (0) -> (e_q, loss) path

  __bf16* Wpack    = (__bf16*)d_ws;                                  // 2 MB packed bf16
  float*  partials = (float*)((char*)d_ws + (size_t)NCB * DIM * 2);  // 256 f32

  float* out_eq = (float*)d_out;                 // [TOK, DIM]
  float* loss   = out_eq + (size_t)TOK * DIM;    // scalar slot

  vq_pack_kernel<<<NTILES, THREADS, 0, stream>>>(W, Wpack);
  vq_sim_argmax_kernel<<<TOK / ROWS_PER_BLOCK, THREADS, 0, stream>>>(
      e, W, Wpack, out_eq, partials);
  vq_finalize_kernel<<<1, THREADS, 0, stream>>>(partials, loss);
}
